// GAT_52140902973477
// MI455X (gfx1250) — matrix operand order repaired and checked by reading, not compiled, not run on hardware
//
#include <hip/hip_runtime.h>
#include <hip/hip_bf16.h>

// ---------------------------------------------------------------------------
// GAT (3x GATConv + linear head) for MI455X / gfx1250.
//  - GEMMs via v_wmma_f32_16x16x32_bf16 (wave32, one 16x16 tile per wave).
//  - Segment softmax via ordered-int atomicMax + atomicAdd denom.
//  - Aggregation: float4 gathers + fp32 atomic scatter-add (bandwidth bound).
// ---------------------------------------------------------------------------

#define N_NODES 100000
#define N_EDGESK 1600000
#define HEADS 4

typedef __attribute__((ext_vector_type(16))) __bf16 bf16x16;
typedef __attribute__((ext_vector_type(8)))  float  f32x8;

// ----- helpers -------------------------------------------------------------

__device__ __forceinline__ int f2ord(float f) {
    int b = __float_as_int(f);
    return b >= 0 ? b : (b ^ 0x7fffffff);   // monotonic float->int mapping
}
__device__ __forceinline__ float ord2f(int b) {
    return __int_as_float(b >= 0 ? b : (b ^ 0x7fffffff));
}
__device__ __forceinline__ void edge_sd(const int* __restrict__ ei, int E, int eid,
                                        int& s, int& d) {
    if (eid < E) { s = ei[eid]; d = ei[E + eid]; }
    else         { s = d = eid - E; }        // appended self-loops
}

// ----- setup kernels -------------------------------------------------------

__global__ void k_cast_bf16(const float* __restrict__ src, __bf16* __restrict__ dst, int n) {
    int i = blockIdx.x * blockDim.x + threadIdx.x;
    if (i < n) dst[i] = (__bf16)src[i];
}

// W[K][Ncols] fp32 -> Wt[Ncols][K] bf16  (so B fragments are contiguous loads)
__global__ void k_transpose_w(const float* __restrict__ W, __bf16* __restrict__ Wt,
                              int K, int Ncols) {
    int i = blockIdx.x * blockDim.x + threadIdx.x;       // i = n*K + k
    if (i >= K * Ncols) return;
    int n = i / K, k = i - n * K;
    Wt[i] = (__bf16)W[(size_t)k * Ncols + n];
}

// ----- WMMA GEMM: C[M,Ncols] = A[M,K](bf16,row-major) x Bt[Ncols,K](bf16) ---

__global__ __launch_bounds__(256)
void k_gemm_wmma(const __bf16* __restrict__ A, const __bf16* __restrict__ Bt,
                 float* __restrict__ C, int M, int K, int Ncols) {
    const int lane  = threadIdx.x & 31;
    const int wave  = threadIdx.x >> 5;
    const int mtile = blockIdx.x * 8 + wave;             // 8 waves / block, stacked in M
    const int base_m = mtile * 16;
    if (base_m >= M) return;                             // wave-uniform: EXEC stays full
    const int base_n = blockIdx.y * 16;

    const int half = lane >> 4;                          // 0 / 1
    const int l16  = lane & 15;

    // ISA 7.12.2 layouts: A lane holds row l16, K = half*8+{0..7} and +16;
    // B lane holds col l16, K = half*8+{0..7} and +16  (Bt is N-major so both are
    // two contiguous 16B loads per 32-K step).
    const __bf16* arow = A  + (size_t)(base_m + l16) * K + half * 8;
    const __bf16* brow = Bt + (size_t)(base_n + l16) * K + half * 8;

    f32x8 acc = {};
    for (int k0 = 0; k0 < K; k0 += 32) {
        union { bf16x16 v; uint4 q[2]; } a, b;
        a.q[0] = *reinterpret_cast<const uint4*>(arow + k0);
        a.q[1] = *reinterpret_cast<const uint4*>(arow + k0 + 16);
        b.q[0] = *reinterpret_cast<const uint4*>(brow + k0);
        b.q[1] = *reinterpret_cast<const uint4*>(brow + k0 + 16);
        acc = __builtin_amdgcn_wmma_f32_16x16x32_bf16(
                  false, a.v, false, b.v, (short)0, acc, false, false);
    }
    // D layout: VGPR r -> row base_m + half*8 + r, col base_n + l16
    float* crow = C + (size_t)(base_m + half * 8) * Ncols + base_n + l16;
#pragma unroll
    for (int r = 0; r < 8; ++r) crow[(size_t)r * Ncols] = acc[r];
}

// ----- attention scalars: as[n,h] = <h[n,h,:], a_src[h,:]> ------------------

__global__ void k_attn_alpha(const float* __restrict__ h, const float* __restrict__ a_src,
                             const float* __restrict__ a_dst, float* __restrict__ as_,
                             float* __restrict__ ad_, int N, int H, int C) {
    int i = blockIdx.x * blockDim.x + threadIdx.x;       // over N*H
    if (i >= N * H) return;
    int n = i / H, hh = i - n * H;
    const float* hp  = h + (size_t)n * H * C + hh * C;
    const float* asw = a_src + hh * C;
    const float* adw = a_dst + hh * C;
    float s = 0.f, d = 0.f;
    for (int c = 0; c < C; ++c) { s += hp[c] * asw[c]; d += hp[c] * adw[c]; }
    as_[i] = s; ad_[i] = d;
}

// ----- per-call accumulator init (harness does not re-clear ws) -------------

__global__ void k_init(int* __restrict__ emax_ord, float* __restrict__ denom, int nh,
                       float* __restrict__ agg, int nhc) {
    int i = blockIdx.x * blockDim.x + threadIdx.x;
    if (i < nh) { emax_ord[i] = 0x807FFFFF /* f2ord(-inf) */; denom[i] = 0.f; }
    if (i < nhc) agg[i] = 0.f;
}

// ----- segment softmax, 3 passes over edges --------------------------------

__global__ void k_edge_max(const int* __restrict__ ei, int E, int Etot,
                           const float* __restrict__ as_, const float* __restrict__ ad_,
                           int* __restrict__ emax_ord, int H) {
    int i = blockIdx.x * blockDim.x + threadIdx.x;
    if (i >= Etot * H) return;
    int eid = i / H, hh = i - eid * H;
    int s, d; edge_sd(ei, E, eid, s, d);
    float e = as_[s * H + hh] + ad_[d * H + hh];
    e = e > 0.f ? e : 0.2f * e;                          // leaky_relu(0.2)
    atomicMax(&emax_ord[d * H + hh], f2ord(e));
}

__global__ void k_edge_denom(const int* __restrict__ ei, int E, int Etot,
                             const float* __restrict__ as_, const float* __restrict__ ad_,
                             const int* __restrict__ emax_ord, float* __restrict__ denom, int H) {
    int i = blockIdx.x * blockDim.x + threadIdx.x;
    if (i >= Etot * H) return;
    int eid = i / H, hh = i - eid * H;
    int s, d; edge_sd(ei, E, eid, s, d);
    float e = as_[s * H + hh] + ad_[d * H + hh];
    e = e > 0.f ? e : 0.2f * e;
    float ex = __expf(e - ord2f(emax_ord[d * H + hh]));
    atomicAdd(&denom[d * H + hh], ex);
}

__global__ void k_edge_coef(const int* __restrict__ ei, int E, int Etot,
                            const float* __restrict__ as_, const float* __restrict__ ad_,
                            const int* __restrict__ emax_ord, const float* __restrict__ denom,
                            float* __restrict__ wcoef, int H) {
    int i = blockIdx.x * blockDim.x + threadIdx.x;
    if (i >= Etot * H) return;
    int eid = i / H, hh = i - eid * H;
    int s, d; edge_sd(ei, E, eid, s, d);
    float e = as_[s * H + hh] + ad_[d * H + hh];
    e = e > 0.f ? e : 0.2f * e;
    float ex = __expf(e - ord2f(emax_ord[d * H + hh]));
    wcoef[i] = ex / (denom[d * H + hh] + 1e-16f);
}

// ----- weighted scatter-add: agg[dst] += wcoef * h[src] --------------------

__global__ void k_edge_aggregate(const int* __restrict__ ei, int E, int Etot,
                                 const float* __restrict__ wcoef,
                                 const float* __restrict__ hfeat,
                                 float* __restrict__ agg, int H, int C) {
    const int HCq = (H * C) >> 2;                        // channel quads per edge
    int i = blockIdx.x * blockDim.x + threadIdx.x;       // Etot*HCq <= 108.8M
    if (i >= Etot * HCq) return;
    int eid = i / HCq, q = i - eid * HCq;
    int s, d; edge_sd(ei, E, eid, s, d);
    int hh = (q << 2) / C;
    float w = wcoef[(size_t)eid * H + hh];
    const float4 v = *reinterpret_cast<const float4*>(hfeat + (size_t)s * H * C + (q << 2));
    float* o = agg + (size_t)d * H * C + (q << 2);
    atomicAdd(o + 0, v.x * w);
    atomicAdd(o + 1, v.y * w);
    atomicAdd(o + 2, v.z * w);
    atomicAdd(o + 3, v.w * w);
}

// ----- bias + ELU, emit next-layer bf16 input (or fp32 for last layer) -----

__global__ void k_finalize(const float* __restrict__ agg, const float* __restrict__ bias,
                           __bf16* __restrict__ out_bf, float* __restrict__ out_f,
                           int N, int HC) {
    int i = blockIdx.x * blockDim.x + threadIdx.x;
    if (i >= N * HC) return;
    int c = i % HC;
    float v = agg[i] + bias[c];
    v = v > 0.f ? v : (__expf(v) - 1.f);                 // ELU
    if (out_bf) out_bf[i] = (__bf16)v;
    if (out_f)  out_f[i]  = v;
}

// ----- head: sigmoid(h3 @ linW + linb) -------------------------------------

__global__ void k_head(const float* __restrict__ h3, const float* __restrict__ linW,
                       const float* __restrict__ linb, float* __restrict__ out, int N) {
    int n = blockIdx.x * blockDim.x + threadIdx.x;
    if (n >= N) return;
    float s = linb[0];
#pragma unroll
    for (int c = 0; c < 16; ++c) s += h3[n * 16 + c] * linW[c];
    out[n] = 1.f / (1.f + __expf(-s));
}

// ---------------------------------------------------------------------------

extern "C" void kernel_launch(void* const* d_in, const int* in_sizes, int n_in,
                              void* d_out, int out_size, void* d_ws, size_t ws_size,
                              hipStream_t stream) {
    (void)in_sizes; (void)n_in; (void)out_size; (void)ws_size;

    const float* x    = (const float*)d_in[0];
    const int*   ei   = (const int*)d_in[1];
    /* d_in[2] = edge_attr: ignored (edge_dim=None) */
    const float* W1   = (const float*)d_in[3];
    const float* a1s  = (const float*)d_in[4];
    const float* a1d  = (const float*)d_in[5];
    const float* b1   = (const float*)d_in[6];
    const float* W2   = (const float*)d_in[7];
    const float* a2s  = (const float*)d_in[8];
    const float* a2d  = (const float*)d_in[9];
    const float* b2   = (const float*)d_in[10];
    const float* W3   = (const float*)d_in[11];
    const float* a3s  = (const float*)d_in[12];
    const float* a3d  = (const float*)d_in[13];
    const float* b3   = (const float*)d_in[14];
    const float* linW = (const float*)d_in[15];
    const float* linb = (const float*)d_in[16];
    float* out = (float*)d_out;

    const int N = N_NODES, E = N_EDGESK, Etot = E + N;

    // ----- workspace carve-up (256B aligned) -----
    char* p = (char*)d_ws;
    auto take = [&](size_t bytes) { char* r = p; p += (bytes + 255) & ~(size_t)255; return (void*)r; };
    __bf16* xbf = (__bf16*)take((size_t)N * 256 * 2);     // layer input, bf16
    __bf16* Wt  = (__bf16*)take((size_t)256 * 256 * 2);   // transposed weights, bf16
    float*  h   = (float*) take((size_t)N * 256 * 4);     // GEMM output
    float*  agg = (float*) take((size_t)N * 256 * 4);     // aggregation accumulator
    float*  as_ = (float*) take((size_t)N * HEADS * 4);
    float*  ad_ = (float*) take((size_t)N * HEADS * 4);
    int*    emx = (int*)   take((size_t)N * HEADS * 4);
    float*  den = (float*) take((size_t)N * HEADS * 4);
    float*  wco = (float*) take((size_t)Etot * HEADS * 4);

    auto cdiv = [](long long a, long long b) { return (unsigned)((a + b - 1) / b); };
    const int BT = 256;

    auto run_layer = [&](const float* Wl, const float* asw, const float* adw,
                         const float* bias, int K, int HC, int H, bool bf_out) {
        const int C = HC / H;
        k_transpose_w<<<cdiv((long long)K * HC, BT), BT, 0, stream>>>(Wl, Wt, K, HC);
        dim3 g(cdiv((long long)N, 16 * 8), HC / 16);
        k_gemm_wmma<<<g, 256, 0, stream>>>(xbf, Wt, h, N, K, HC);
        k_attn_alpha<<<cdiv((long long)N * H, BT), BT, 0, stream>>>(h, asw, adw, as_, ad_, N, H, C);
        k_init<<<cdiv((long long)N * HC, BT), BT, 0, stream>>>(emx, den, N * H, agg, N * HC);
        k_edge_max<<<cdiv((long long)Etot * H, BT), BT, 0, stream>>>(ei, E, Etot, as_, ad_, emx, H);
        k_edge_denom<<<cdiv((long long)Etot * H, BT), BT, 0, stream>>>(ei, E, Etot, as_, ad_, emx, den, H);
        k_edge_coef<<<cdiv((long long)Etot * H, BT), BT, 0, stream>>>(ei, E, Etot, as_, ad_, emx, den, wco, H);
        k_edge_aggregate<<<cdiv((long long)Etot * (HC / 4), BT), BT, 0, stream>>>(
            ei, E, Etot, wco, h, agg, H, C);
        k_finalize<<<cdiv((long long)N * HC, BT), BT, 0, stream>>>(
            agg, bias, bf_out ? xbf : nullptr, bf_out ? nullptr : h, N, HC);
    };

    // conv1: 128 -> 4 heads x 64, concat(256), ELU  -> xbf
    k_cast_bf16<<<cdiv((long long)N * 128, BT), BT, 0, stream>>>(x, xbf, N * 128);
    run_layer(W1, a1s, a1d, b1, 128, 256, HEADS, /*bf_out=*/true);
    // conv2: 256 -> 64, 1 head (mean==identity), ELU -> xbf
    run_layer(W2, a2s, a2d, b2, 256, 64, 1, /*bf_out=*/true);
    // conv3: 64 -> 16, 1 head, ELU -> h (fp32)
    run_layer(W3, a3s, a3d, b3, 64, 16, 1, /*bf_out=*/false);
    // head: sigmoid(h @ linW + linb) -> d_out [N]
    k_head<<<cdiv((long long)N, BT), BT, 0, stream>>>(h, linW, linb, out, N);
}
